// SmoothQuantLinear_3753801416785
// MI455X (gfx1250) — 8-bit matrix operand order repaired and checked by reading, compiled, not run on hardware
//
#include <hip/hip_runtime.h>
#include <stdint.h>


typedef __attribute__((ext_vector_type(8))) int   v8i;
typedef __attribute__((ext_vector_type(4))) float v4f;

constexpr int IN_F    = 4096;
constexpr int OUT_F   = 4096;
constexpr int M_TOTAL = 4 * 2048;
constexpr long TOTAL4 = (long)M_TOTAL * IN_F / 4;

constexpr size_t XQ_OFFSET = 256;

__global__ void sq_init(unsigned int* gmax) {
    gmax[0] = 0u;
}

__global__ __launch_bounds__(256)
void sq_absmax(const float* __restrict__ x, const float* __restrict__ smooth,
               unsigned int* __restrict__ gmax) {
    __shared__ float red[256];
    float m = 0.0f;
    const long stride = (long)gridDim.x * blockDim.x;
    for (long i = (long)blockIdx.x * blockDim.x + threadIdx.x; i < TOTAL4; i += stride) {
        v4f xv = ((const v4f*)x)[i];
        v4f sv = ((const v4f*)smooth)[i & (IN_F / 4 - 1)];
        m = fmaxf(m, fabsf(xv.x) / sv.x);
        m = fmaxf(m, fabsf(xv.y) / sv.y);
        m = fmaxf(m, fabsf(xv.z) / sv.z);
        m = fmaxf(m, fabsf(xv.w) / sv.w);
    }
    red[threadIdx.x] = m;
    __syncthreads();
    #pragma unroll
    for (int s = 128; s > 0; s >>= 1) {
        if (threadIdx.x < s)
            red[threadIdx.x] = fmaxf(red[threadIdx.x], red[threadIdx.x + s]);
        __syncthreads();
    }
    if (threadIdx.x == 0)
        atomicMax(gmax, __float_as_uint(red[0]));
}

__global__ __launch_bounds__(256)
void sq_quant(const float* __restrict__ x, const float* __restrict__ smooth,
              const unsigned int* __restrict__ gmax,
              unsigned int* __restrict__ xq_packed, float* __restrict__ act_scale_out) {
    const float amax  = __uint_as_float(*gmax);
    const float scale = amax * (1.0f / 127.0f);
    const float inv   = 1.0f / (scale + 1e-5f);
    if (blockIdx.x == 0 && threadIdx.x == 0) *act_scale_out = scale;

    const long stride = (long)gridDim.x * blockDim.x;
    for (long i = (long)blockIdx.x * blockDim.x + threadIdx.x; i < TOTAL4; i += stride) {
        v4f xv = ((const v4f*)x)[i];
        v4f sv = ((const v4f*)smooth)[i & (IN_F / 4 - 1)];
        int q0 = (int)rintf(fminf(fmaxf(xv.x / sv.x * inv, -128.0f), 127.0f));
        int q1 = (int)rintf(fminf(fmaxf(xv.y / sv.y * inv, -128.0f), 127.0f));
        int q2 = (int)rintf(fminf(fmaxf(xv.z / sv.z * inv, -128.0f), 127.0f));
        int q3 = (int)rintf(fminf(fmaxf(xv.w / sv.w * inv, -128.0f), 127.0f));
        unsigned int p = ((unsigned)q0 & 0xFF)        | (((unsigned)q1 & 0xFF) << 8) |
                         (((unsigned)q2 & 0xFF) << 16) | (((unsigned)q3 & 0xFF) << 24);
        xq_packed[i] = p;
    }
}

__global__ __launch_bounds__(256)
void sq_gemm(const int8_t* __restrict__ xq, const int8_t* __restrict__ w,
             const float* __restrict__ wscale, const float* __restrict__ bias,
             const float* __restrict__ act_scale_p, float* __restrict__ out) {
    const int lane = threadIdx.x & 31;
    const int wave = threadIdx.x >> 5;
    const int r = lane & 15;
    const int h = lane >> 4;
    const int mw = wave & 1;
    const int nw = wave >> 1;
    const int m_base = blockIdx.y * 64  + mw * 32;
    const int n_base = blockIdx.x * 256 + nw * 64;

    v8i acc[2][4];
    #pragma unroll
    for (int t = 0; t < 2; ++t)
        #pragma unroll
        for (int u = 0; u < 4; ++u)
            acc[t][u] = (v8i){0, 0, 0, 0, 0, 0, 0, 0};

    const int8_t* arow0 = xq + (long)(m_base + r)      * IN_F;
    const int8_t* arow1 = xq + (long)(m_base + 16 + r) * IN_F;
    const int8_t* brow[4];
    #pragma unroll
    for (int u = 0; u < 4; ++u)
        brow[u] = w + (long)(n_base + u * 16 + r) * IN_F;

    for (int k0 = 0; k0 < IN_F; k0 += 64) {
        union { v8i v; uint64_t d[4]; } ua[2];
        {
            const uint64_t* p0 = (const uint64_t*)(arow0 + k0 + h * 8);
            const uint64_t* p1 = (const uint64_t*)(arow1 + k0 + h * 8);
            #pragma unroll
            for (int g = 0; g < 4; ++g) {
                ua[0].d[g] = p0[2 * g];
                ua[1].d[g] = p1[2 * g];
            }
        }
        union { v8i v; uint64_t d[4]; } ub[4];
        #pragma unroll
        for (int u = 0; u < 4; ++u) {
            const uint64_t* p = (const uint64_t*)(brow[u] + k0 + h * 8);
            ub[u].d[0] = p[0];
            ub[u].d[1] = p[2];
            ub[u].d[2] = p[4];
            ub[u].d[3] = p[6];
        }
        #pragma unroll
        for (int t = 0; t < 2; ++t)
            #pragma unroll
            for (int u = 0; u < 4; ++u)
                acc[t][u] = __builtin_amdgcn_wmma_i32_16x16x64_iu8(
                    true, (t == 0 ? ua[0].v : ua[1].v),
                    true, ub[u].v,
                    acc[t][u], false, false);
    }

    const float act_scale = *act_scale_p;
    #pragma unroll
    for (int u = 0; u < 4; ++u) {
        const int n = n_base + u * 16 + r;
        const float sc = wscale[n] * act_scale;
        const float bb = bias[n];
        #pragma unroll
        for (int t = 0; t < 2; ++t) {
            const int m0 = m_base + t * 16 + 8 * h;
            #pragma unroll
            for (int j = 0; j < 8; ++j) {
                out[(long)(m0 + j) * OUT_F + n] = (float)acc[t][u][j] * sc + bb;
            }
        }
    }
}

extern "C" void kernel_launch(void* const* d_in, const int* in_sizes, int n_in,
                              void* d_out, int out_size, void* d_ws, size_t ws_size,
                              hipStream_t stream) {
    const float*  x      = (const float*)d_in[0];
    const int8_t* wq     = (const int8_t*)d_in[1];
    const float*  wscale = (const float*)d_in[2];
    const float*  smooth = (const float*)d_in[3];
    const float*  bias   = (const float*)d_in[4];
    float*        out    = (float*)d_out;

    unsigned int* gmax      = (unsigned int*)d_ws;
    float*        act_scale = (float*)((char*)d_ws + 4);
    int8_t*       xq        = (int8_t*)((char*)d_ws + XQ_OFFSET);

    sq_init<<<1, 1, 0, stream>>>(gmax);
    sq_absmax<<<2048, 256, 0, stream>>>(x, smooth, gmax);
    sq_quant<<<4096, 256, 0, stream>>>(x, smooth, gmax, (unsigned int*)xq, act_scale);

    dim3 grid(OUT_F / 256, M_TOTAL / 64);
    sq_gemm<<<grid, 256, 0, stream>>>(xq, wq, wscale, bias, act_scale, out);
}
